// MultiHeadAttention_45664092291153
// MI455X (gfx1250) — compile-verified, matrix operand order checked by reading, not run on hardware
//
#include <hip/hip_runtime.h>

typedef _Float16 v8h  __attribute__((ext_vector_type(8)));
typedef _Float16 v16h __attribute__((ext_vector_type(16)));
typedef float    v8f  __attribute__((ext_vector_type(8)));
typedef _Float16 h4   __attribute__((ext_vector_type(4)));
typedef int      v4i  __attribute__((ext_vector_type(4)));
typedef int      v8i  __attribute__((ext_vector_type(8)));

#define B_  2
#define S_  2048
#define D_  1024
#define H_  16
#define HD_ 64
#define M_  (B_*S_)      /* 4096 rows of the fused (B*S) dimension */
#define SCALE_ 0.125f    /* 1/sqrt(64) */

union Frag { v16h v; v8h h[2]; };

// 16x32 f16 WMMA operand fragment (CDNA5 16-bit A layout; B assumed mirrored):
// lane L (L&15 = row) holds K = {hi*8..hi*8+7} and {16+hi*8..16+hi*8+7}.
__device__ __forceinline__ v16h load_frag(const _Float16* rowchunk, int hi) {
  Frag f;
  f.h[0] = *(const v8h*)(rowchunk + hi * 8);
  f.h[1] = *(const v8h*)(rowchunk + 16 + hi * 8);
  return f.v;
}

__device__ __forceinline__ v8f wmma_f16(v16h a, v16h b, v8f c) {
  return __builtin_amdgcn_wmma_f32_16x16x32_f16(false, a, false, b,
                                                (short)0, c, false, false);
}

__device__ __forceinline__ int rfl(unsigned v) {
  return __builtin_amdgcn_readfirstlane((int)v);
}

// ---------------------------------------------------------------------------
// Tensor Data Mover: async 2D f16 tile load (global -> LDS), ISA §8.3/8.4.
// All dims/strides in elements (data_size = 1 -> 2 bytes). Tile assumed fully
// in-bounds, so tensor dims are set to the tile dims. Tracked by TENSORcnt.
// ---------------------------------------------------------------------------
__device__ __forceinline__ void tdm_load_2d_f16(unsigned lds_byte_off,
                                                const void* gptr,
                                                unsigned len0, unsigned len1,
                                                unsigned stride0,
                                                unsigned tile0, unsigned tile1) {
  const unsigned long long ga = (unsigned long long)(uintptr_t)gptr;
  v4i g0;
  g0.x = rfl(1u);                                  // count=1, flags=0
  g0.y = rfl(lds_byte_off);                        // lds_addr (bytes)
  g0.z = rfl((unsigned)ga);                        // global_addr[31:0]
  g0.w = rfl(((unsigned)(ga >> 32) & 0x01FFFFFFu) | (2u << 30)); // [56:32]|type=2
  v8i g1;
  g1.s0 = rfl(1u << 16);                           // data_size=1 (2B), mask=0
  g1.s1 = rfl((len0 & 0xFFFFu) << 16);             // tensor_dim0[15:0]
  g1.s2 = rfl((len0 >> 16) | ((len1 & 0xFFFFu) << 16));
  g1.s3 = rfl((len1 >> 16) | (tile0 << 16));       // tile_dim0
  g1.s4 = rfl(tile1);                              // tile_dim1 (tile_dim2=0)
  g1.s5 = rfl(stride0);                            // tensor_dim0_stride[31:0]
  g1.s6 = rfl(0u);                                 // stride hi, dim1_stride lo
  g1.s7 = rfl(0u);
  asm volatile("tensor_load_to_lds %0, %1" :: "s"(g0), "s"(g1) : "memory");
}

// ---------------------------------------------------------------------------
// fp32 -> fp16 cast (hidden state + weights). n is a multiple of 4.
// ---------------------------------------------------------------------------
__global__ __launch_bounds__(256) void cast_f32_f16(const float* __restrict__ in,
                                                    _Float16* __restrict__ out,
                                                    int n) {
  int i = (blockIdx.x * blockDim.x + threadIdx.x) * 4;
  if (i >= n) return;
  float4 v = *(const float4*)(in + i);
  h4 o = { (_Float16)v.x, (_Float16)v.y, (_Float16)v.z, (_Float16)v.w };
  *(h4*)(out + i) = o;
}

// ---------------------------------------------------------------------------
// GEMM: C[m,n] = sum_k A[m,k]*W[n,k] + bias[n]  (X @ W^T + b), all K = D_.
// Block = 8 waves covering a 32(m) x 256(n) tile; per 64-wide k-chunk the TDM
// stages A(32x64) + W(256x64) into double-buffered LDS, overlapped with WMMA
// on the previous chunk (s_wait_tensorcnt <= 2 releases chunk c while c+1 is
// still in flight). Each wave owns 32x32 of C (2x2 accumulator tiles).
// mode 0: f16 out, Q/K head layout [B,H,S,HD]
// mode 2: f16 out, transposed V layout [B,H,HD,S]
// mode 3: f32 out, row-major [M_, D_]
// ---------------------------------------------------------------------------
#define GK_CH 64
#define GK_NCH (D_ / GK_CH)   /* 16 */

__global__ __launch_bounds__(256) void gemm_wmma(const _Float16* __restrict__ A,
                                                 const _Float16* __restrict__ W,
                                                 const float* __restrict__ bias,
                                                 void* __restrict__ out,
                                                 int mode) {
  __shared__ __align__(16) _Float16 Asm[2][32 * GK_CH];    //  8 KB
  __shared__ __align__(16) _Float16 Wsm[2][256 * GK_CH];   // 64 KB

  const int lane = threadIdx.x & 31;
  const int w    = threadIdx.x >> 5;
  const int row  = lane & 15;
  const int hi   = lane >> 4;

  const int mt = blockIdx.x >> 2;          // 128 m-tiles of 32 rows
  const int nb = blockIdx.x & 3;           // 4 n-panels of 256 cols
  const int m0 = mt * 32;
  const int n0base = nb * 256;
  const int n0 = n0base + w * 32;

  const _Float16* Ag = A + (size_t)m0 * D_;
  const _Float16* Wg = W + (size_t)n0base * D_;

  if (w == 0) {
    tdm_load_2d_f16((unsigned)(uintptr_t)&Asm[0][0], Ag, GK_CH, 32, D_, GK_CH, 32);
    tdm_load_2d_f16((unsigned)(uintptr_t)&Wsm[0][0], Wg, GK_CH, 256, D_, GK_CH, 256);
  }

  v8f acc00 = {}, acc01 = {}, acc10 = {}, acc11 = {};

  for (int c = 0; c < GK_NCH; ++c) {
    if (w == 0) {
      if (c + 1 < GK_NCH) {
        const int nb2 = (c + 1) & 1;
        tdm_load_2d_f16((unsigned)(uintptr_t)&Asm[nb2][0], Ag + (c + 1) * GK_CH,
                        GK_CH, 32, D_, GK_CH, 32);
        tdm_load_2d_f16((unsigned)(uintptr_t)&Wsm[nb2][0], Wg + (c + 1) * GK_CH,
                        GK_CH, 256, D_, GK_CH, 256);
        __builtin_amdgcn_s_wait_tensorcnt(2);   // chunk c finished
      } else {
        __builtin_amdgcn_s_wait_tensorcnt(0);
      }
    }
    __syncthreads();

    const _Float16* As = &Asm[c & 1][0];
    const _Float16* Ws = &Wsm[c & 1][0] + (size_t)(w * 32) * GK_CH;
    #pragma unroll
    for (int ks = 0; ks < GK_CH; ks += 32) {
      v16h fa0 = load_frag(As + (size_t)row * GK_CH + ks, hi);
      v16h fa1 = load_frag(As + (size_t)(16 + row) * GK_CH + ks, hi);
      v16h fw0 = load_frag(Ws + (size_t)row * GK_CH + ks, hi);
      v16h fw1 = load_frag(Ws + (size_t)(16 + row) * GK_CH + ks, hi);
      acc00 = wmma_f16(fa0, fw0, acc00);
      acc01 = wmma_f16(fa0, fw1, acc01);
      acc10 = wmma_f16(fa1, fw0, acc10);
      acc11 = wmma_f16(fa1, fw1, acc11);
    }
    __syncthreads();
  }

  const float bv0 = bias[n0 + row];
  const float bv1 = bias[n0 + 16 + row];
  v8f acc[2][2] = { { acc00, acc01 }, { acc10, acc11 } };

  #pragma unroll
  for (int i = 0; i < 2; ++i) {
    #pragma unroll
    for (int j = 0; j < 2; ++j) {
      #pragma unroll
      for (int r = 0; r < 8; ++r) {
        const float val = acc[i][j][r] + (j ? bv1 : bv0);
        const int m = m0 + i * 16 + r + hi * 8;   // C layout: VGPR r -> row r/r+8
        const int n = n0 + j * 16 + row;          // lane -> column
        if (mode == 3) {
          ((float*)out)[(size_t)m * D_ + n] = val;
        } else {
          const int q = m & (S_ - 1), b = m >> 11;   // S_ == 2048
          const int h = n >> 6, hd = n & (HD_ - 1);
          size_t addr;
          if (mode == 2)  // V transposed: [B,H,HD,S]
            addr = ((size_t)(b * H_ + h) * HD_ + hd) * S_ + q;
          else            // Q/K: [B,H,S,HD]
            addr = ((size_t)(b * H_ + h) * S_ + q) * HD_ + hd;
          ((_Float16*)out)[addr] = (_Float16)val;
        }
      }
    }
  }
}

// ---------------------------------------------------------------------------
// Flash attention, transposed-score formulation.  Block = 8 waves, one (b,h),
// 8 consecutive 16-query tiles; K/V panels staged per 128-key chunk by the TDM
// into double-buffered LDS and shared by all waves.  S'[k,q] = K·Qᵀ/8+mask[k];
// online softmax over k (lane == q, one shfl_xor(16) per reduction); exp(S')
// in the C register layout *is* the next WMMA's B operand (no shuffles);
// O'[d,q] += Vᵀ·P' with V pre-transposed in memory by the V-projection GEMM.
// ---------------------------------------------------------------------------
#define AT_CH 128
#define AT_NCH (S_ / AT_CH)   /* 16 */

__global__ __launch_bounds__(256) void attn_wmma(const _Float16* __restrict__ Q,
                                                 const _Float16* __restrict__ K,
                                                 const _Float16* __restrict__ Vt,
                                                 const float* __restrict__ mask,
                                                 _Float16* __restrict__ ctx) {
  __shared__ __align__(16) _Float16 Ksm[2][AT_CH * HD_];   // 32 KB (rows x 64)
  __shared__ __align__(16) _Float16 Vsm[2][HD_ * AT_CH];   // 32 KB (64 x cols)

  const int lane = threadIdx.x & 31;
  const int w    = threadIdx.x >> 5;
  const int row  = lane & 15;
  const int hi   = lane >> 4;

  const int bid = blockIdx.x;
  const int qb  = bid & 15;                 // 16 query panels of 128
  const int h   = (bid >> 4) & (H_ - 1);
  const int b   = bid >> 8;
  const int q0  = qb * 128 + w * 16;

  const _Float16* Qp = Q  + ((size_t)(b * H_ + h) * S_ + q0) * HD_;
  const _Float16* Kp = K  + ((size_t)(b * H_ + h) * S_) * HD_;
  const _Float16* Vp = Vt + ((size_t)(b * H_ + h) * HD_) * S_;
  const float*    mp = mask + (size_t)b * S_;

  // Q as B operand (lane = q column), held for the whole key loop.
  v16h bq0 = load_frag(Qp + row * HD_ + 0,  hi);
  v16h bq1 = load_frag(Qp + row * HD_ + 32, hi);

  if (w == 0) {
    tdm_load_2d_f16((unsigned)(uintptr_t)&Ksm[0][0], Kp, HD_, AT_CH, HD_, HD_, AT_CH);
    tdm_load_2d_f16((unsigned)(uintptr_t)&Vsm[0][0], Vp, AT_CH, HD_, S_, AT_CH, HD_);
  }

  v8f O[4] = {};                 // O'[d,q]: 4 tiles cover d = 0..63
  float mrun = -1e30f, lrun = 0.f;
  const int mskbase = (lane & 16) >> 1;   // 0 for lanes 0-15, 8 for 16-31

  for (int c = 0; c < AT_NCH; ++c) {
    if (w == 0) {
      if (c + 1 < AT_NCH) {
        const int nb2 = (c + 1) & 1;
        tdm_load_2d_f16((unsigned)(uintptr_t)&Ksm[nb2][0],
                        Kp + (size_t)(c + 1) * AT_CH * HD_,
                        HD_, AT_CH, HD_, HD_, AT_CH);
        tdm_load_2d_f16((unsigned)(uintptr_t)&Vsm[nb2][0],
                        Vp + (size_t)(c + 1) * AT_CH,
                        AT_CH, HD_, S_, AT_CH, HD_);
        __builtin_amdgcn_s_wait_tensorcnt(2);   // chunk c landed
      } else {
        __builtin_amdgcn_s_wait_tensorcnt(0);
      }
    }
    __syncthreads();

    const _Float16* Ks = &Ksm[c & 1][0];
    const _Float16* Vs = &Vsm[c & 1][0];

    #pragma unroll
    for (int kbl = 0; kbl < AT_CH; kbl += 32) {
      // scores for 32 keys: two 16x16 tiles (k = kbl.., kbl+16..)
      v8f sa = {}, sb = {};
      {
        v16h ka0 = load_frag(Ks + (size_t)(kbl + row) * HD_ + 0,  hi);
        v16h ka1 = load_frag(Ks + (size_t)(kbl + row) * HD_ + 32, hi);
        sa = wmma_f16(ka0, bq0, sa);
        sa = wmma_f16(ka1, bq1, sa);
        v16h kb0 = load_frag(Ks + (size_t)(kbl + 16 + row) * HD_ + 0,  hi);
        v16h kb1 = load_frag(Ks + (size_t)(kbl + 16 + row) * HD_ + 32, hi);
        sb = wmma_f16(kb0, bq0, sb);
        sb = wmma_f16(kb1, bq1, sb);
      }

      // scale + additive mask (mask indexed by key row -> broadcast via shfl)
      const float mv = mp[c * AT_CH + kbl + lane];
      float pa[8], pb[8];
      float bm = -1e30f;
      #pragma unroll
      for (int r = 0; r < 8; ++r) {
        pa[r] = sa[r] * SCALE_ + __shfl(mv, mskbase + r, 32);
        pb[r] = sb[r] * SCALE_ + __shfl(mv, 16 + mskbase + r, 32);
        bm = fmaxf(bm, fmaxf(pa[r], pb[r]));
      }
      bm = fmaxf(bm, __shfl_xor(bm, 16, 32));   // combine half-waves (same q)

      const float mnew = fmaxf(mrun, bm);
      const float corr = __expf(mrun - mnew);
      mrun = mnew;

      // exponentiate; C layout packs directly into the next WMMA's B operand
      float ps = 0.f;
      Frag bp;
      #pragma unroll
      for (int r = 0; r < 8; ++r) {
        pa[r] = __expf(pa[r] - mnew);  ps += pa[r];
        pb[r] = __expf(pb[r] - mnew);  ps += pb[r];
        bp.v[r]     = (_Float16)pa[r];
        bp.v[8 + r] = (_Float16)pb[r];
      }
      ps += __shfl_xor(ps, 16, 32);
      lrun = lrun * corr + ps;

      #pragma unroll
      for (int t = 0; t < 4; ++t) {
        #pragma unroll
        for (int r = 0; r < 8; ++r) O[t][r] *= corr;
      }

      // O'[d,q] += Vᵀ(16d x 32k) · P'(32k x 16q); Vᵀ rows contiguous in LDS
      #pragma unroll
      for (int t = 0; t < 4; ++t) {
        const _Float16* vp = Vs + (size_t)(t * 16 + row) * AT_CH + kbl;
        Frag av;
        av.h[0] = *(const v8h*)(vp + hi * 8);
        av.h[1] = *(const v8h*)(vp + 16 + hi * 8);
        O[t] = wmma_f16(av.v, bp.v, O[t]);
      }
    }
    __syncthreads();
  }

  // normalize and store ctx[b, q, h*64 + d] (d = VGPR row, q = lane column)
  const float inv = 1.f / lrun;
  const int q = q0 + row;
  #pragma unroll
  for (int t = 0; t < 4; ++t) {
    #pragma unroll
    for (int r = 0; r < 8; ++r) {
      const int d = t * 16 + r + hi * 8;
      ctx[((size_t)b * S_ + q) * D_ + h * HD_ + d] = (_Float16)(O[t][r] * inv);
    }
  }
}

// ---------------------------------------------------------------------------
extern "C" void kernel_launch(void* const* d_in, const int* in_sizes, int n_in,
                              void* d_out, int out_size, void* d_ws, size_t ws_size,
                              hipStream_t stream) {
  (void)in_sizes; (void)n_in; (void)out_size; (void)ws_size;
  const float* hs   = (const float*)d_in[0];
  const float* mask = (const float*)d_in[1];
  const float* Wq = (const float*)d_in[2];  const float* bq = (const float*)d_in[3];
  const float* Wk = (const float*)d_in[4];  const float* bk = (const float*)d_in[5];
  const float* Wv = (const float*)d_in[6];  const float* bv = (const float*)d_in[7];
  const float* Wo = (const float*)d_in[8];  const float* bo = (const float*)d_in[9];

  char* ws = (char*)d_ws;
  size_t off = 0;
  auto alloc = [&](size_t bytes) -> char* {
    char* p = ws + off;
    off += (bytes + 255) & ~(size_t)255;
    return p;
  };
  _Float16* Xh   = (_Float16*)alloc((size_t)M_ * D_ * 2);   // hidden f16
  _Float16* Wq16 = (_Float16*)alloc((size_t)D_ * D_ * 2);
  _Float16* Wk16 = (_Float16*)alloc((size_t)D_ * D_ * 2);
  _Float16* Wv16 = (_Float16*)alloc((size_t)D_ * D_ * 2);
  _Float16* Wo16 = (_Float16*)alloc((size_t)D_ * D_ * 2);
  _Float16* Qh   = (_Float16*)alloc((size_t)M_ * D_ * 2);   // [B,H,S,HD]
  _Float16* Kh   = (_Float16*)alloc((size_t)M_ * D_ * 2);   // [B,H,S,HD]
  _Float16* Vth  = (_Float16*)alloc((size_t)M_ * D_ * 2);   // [B,H,HD,S]
  _Float16* Ctx  = (_Float16*)alloc((size_t)M_ * D_ * 2);   // [B,S,D]

  cast_f32_f16<<<(M_ * D_) / 1024, 256, 0, stream>>>(hs, Xh, M_ * D_);
  cast_f32_f16<<<(D_ * D_) / 1024, 256, 0, stream>>>(Wq, Wq16, D_ * D_);
  cast_f32_f16<<<(D_ * D_) / 1024, 256, 0, stream>>>(Wk, Wk16, D_ * D_);
  cast_f32_f16<<<(D_ * D_) / 1024, 256, 0, stream>>>(Wv, Wv16, D_ * D_);
  cast_f32_f16<<<(D_ * D_) / 1024, 256, 0, stream>>>(Wo, Wo16, D_ * D_);

  const int gemmBlocks = 128 * 4;   // 128 m-tiles x 4 n-panels
  gemm_wmma<<<gemmBlocks, 256, 0, stream>>>(Xh, Wq16, bq, Qh, 0);
  gemm_wmma<<<gemmBlocks, 256, 0, stream>>>(Xh, Wk16, bk, Kh, 0);
  gemm_wmma<<<gemmBlocks, 256, 0, stream>>>(Xh, Wv16, bv, Vth, 2);

  attn_wmma<<<B_ * H_ * (S_ / 128), 256, 0, stream>>>(Qh, Kh, Vth, mask, Ctx);

  gemm_wmma<<<gemmBlocks, 256, 0, stream>>>(Ctx, Wo16, bo, d_out, 3);
}
